// RecurrentLayer_56049323213026
// MI455X (gfx1250) — matrix operand order repaired and checked by reading, not compiled, not run on hardware
//
#include <hip/hip_runtime.h>
#include <hip/hip_bf16.h>

// Problem dims (fixed by the reference)
#define B_DIM 64
#define T_DIM 1024
#define D_DIM 512
#define H_DIM 512

// Scan kernel tiling: 4 independent WGs (16 batch rows each), 16 waves/WG.
// Wh split: K-blocks 0..KB_LDS-1 in LDS, KB_LDS..15 in registers.
// LDS: 256 KB Wh + 2 x 16 KB double-buffered h state = 288 KB (of 320 KB).
#define KB_TOT 16
#define KB_LDS 8
#define KB_REG (KB_TOT - KB_LDS)
#define HBUF_ELEMS (KB_TOT * 32 * 16)                       // 8192 bf16
#define SCAN_LDS_BYTES ((KB_LDS * 32 * 32 * 16 + 2 * HBUF_ELEMS) * 2)

typedef __attribute__((ext_vector_type(16))) __bf16 v16bf;
typedef __attribute__((ext_vector_type(8)))  float  v8f;

// ---------------------------------------------------------------------------
// Fast tanh: native V_TANH_F32 if the builtin exists, else exp2+rcp (3 ops,
// saturates to +/-1, propagates NaN).
// ---------------------------------------------------------------------------
static __device__ __forceinline__ float fast_tanh(float x) {
#if __has_builtin(__builtin_amdgcn_tanhf)
  return __builtin_amdgcn_tanhf(x);
#elif __has_builtin(__builtin_amdgcn_tanh_f32)
  return __builtin_amdgcn_tanh_f32(x);
#else
  // tanh(x) = 1 - 2/(exp(2x)+1);  2*log2(e) = 2.8853900817779268
  const float e = __builtin_amdgcn_exp2f(2.885390082f * x);
  return 1.0f - 2.0f * __builtin_amdgcn_rcpf(e + 1.0f);
#endif
}

// ---------------------------------------------------------------------------
// A-fragment loader from fp32 memory (used by phase 1 only).
// A layout (contiguous k): lanes 0-15: M=lane, K=0..15;
//                         lanes 16-31: M=lane-16, K=16..31.
// ---------------------------------------------------------------------------
static __device__ __forceinline__ v16bf load_a_frag_f32(const float* rowPtr,
                                                        int k0, int lane) {
  const int kbase = (lane >= 16) ? 16 : 0;
  const float* p = rowPtr + k0 + kbase;
  const float4 f0 = *(const float4*)(p + 0);    // K+0..3
  const float4 f1 = *(const float4*)(p + 4);    // K+4..7
  const float4 f2 = *(const float4*)(p + 8);    // K+8..11
  const float4 f3 = *(const float4*)(p + 12);   // K+12..15
  v16bf a;
  a[0]  = (__bf16)f0.x; a[1]  = (__bf16)f0.y; a[2]  = (__bf16)f0.z; a[3]  = (__bf16)f0.w;
  a[4]  = (__bf16)f1.x; a[5]  = (__bf16)f1.y; a[6]  = (__bf16)f1.z; a[7]  = (__bf16)f1.w;
  a[8]  = (__bf16)f2.x; a[9]  = (__bf16)f2.y; a[10] = (__bf16)f2.z; a[11] = (__bf16)f2.w;
  a[12] = (__bf16)f3.x; a[13] = (__bf16)f3.y; a[14] = (__bf16)f3.z; a[15] = (__bf16)f3.w;
  return a;
}

// ---------------------------------------------------------------------------
// Phase 1: xw = x @ Wx + b  ->  d_out.   grid=(32 colblocks, 64), block=128.
// Wx column slice staged in LDS as bf16 B fragments.
// ---------------------------------------------------------------------------
__global__ __launch_bounds__(128) void rnn_xw_gemm(const float* __restrict__ x,
                                                   const float* __restrict__ W,
                                                   const float* __restrict__ bias,
                                                   float* __restrict__ out) {
  __shared__ __bf16 Bsh[16 * 32 * 16];   // 16 KB: [kb][lane][e]
  const int colBase = blockIdx.x * 16;
  const int tid  = threadIdx.x;
  const int lane = tid & 31;
  const int wave = tid >> 5;

  // B layout (32x16): lanes 0-15 -> K=k0..k0+15, lanes 16-31 -> K+16..31.
  for (int idx = tid; idx < 16 * 32 * 16; idx += 128) {
    const int kb  = idx >> 9;
    const int l   = (idx >> 4) & 31;
    const int e   = idx & 15;
    const int K   = kb * 32 + ((l >= 16) ? 16 : 0) + e;
    Bsh[idx] = (__bf16)W[(size_t)K * H_DIM + colBase + (l & 15)];
  }
  __syncthreads();

  const int   n    = colBase + (lane & 15);
  const float bcol = bias[n];
  const int   mOff = (lane >= 16) ? 8 : 0;
  const size_t BT  = (size_t)B_DIM * T_DIM;

  for (size_t mSuper = blockIdx.y; mSuper < BT / 64; mSuper += gridDim.y) {
    const size_t rowBase = mSuper * 64 + (size_t)wave * 16;
    const float* rowPtr  = x + (rowBase + (size_t)(lane & 15)) * D_DIM;

    v8f acc = {0.f, 0.f, 0.f, 0.f, 0.f, 0.f, 0.f, 0.f};
#pragma unroll
    for (int kb = 0; kb < 16; ++kb) {
      v16bf a = load_a_frag_f32(rowPtr, kb * 32, lane);
      v16bf b = *(const v16bf*)&Bsh[(size_t)(kb * 32 + lane) * 16];
      acc = __builtin_amdgcn_wmma_f32_16x16x32_bf16(false, a, false, b,
                                                    (short)0, acc, false, false);
    }
#pragma unroll
    for (int r = 0; r < 8; ++r) {
      out[(rowBase + mOff + r) * H_DIM + n] = acc[r] + bcol;
    }
  }
}

// ---------------------------------------------------------------------------
// Phase 2: barrier-free recurrent scan.
// grid = 4 WGs (one per 16-row batch block), block = 512 (16 waves).
// Each WG computes ALL 512 h-columns for its rows, so the recurrence never
// leaves the WGP.  h state: double-buffered bf16 LDS in WMMA A-fragment
// layout (one __syncthreads per step).  Wh: K<256 in LDS B-fragment layout,
// K>=256 in per-wave VGPRs (2 col blocks x 8 kb x 8 VGPRs = 128 VGPRs).
// ---------------------------------------------------------------------------
__global__ __launch_bounds__(512) void rnn_scan_local(const float* __restrict__ W,
                                                      float* out) {
  extern __shared__ char smem_raw[];
  __bf16* WhL  = (__bf16*)smem_raw;                 // [KB_LDS][nb 32][lane 32][e 16]
  __bf16* hbuf = WhL + KB_LDS * 32 * 32 * 16;       // [2][kb 16][lane 32][e 16]

  const int tid  = threadIdx.x;
  const int lane = tid & 31;
  const int wave = tid >> 5;
  const int rowB = blockIdx.x * 16;          // this WG's batch rows

  // ---- stage Wh K-blocks 0..KB_LDS-1 (all 512 cols) into LDS as bf16 ----
  for (int idx = tid; idx < KB_LDS * 32 * 32 * 16; idx += 512) {
    const int e  = idx & 15;
    const int l  = (idx >> 4) & 31;
    const int nb = (idx >> 9) & 31;
    const int kb = idx >> 14;
    const int K  = kb * 32 + ((l >= 16) ? 8 : 0) + e + ((e >= 8) ? 8 : 0);
    WhL[idx] = (__bf16)W[(size_t)(D_DIM + K) * H_DIM + nb * 16 + (l & 15)];
  }

  // ---- per-wave register copy of Wh K-blocks KB_LDS..15, 2 col blocks ----
  const int nb0 = wave * 2;                  // this wave's column blocks
  v16bf whi[2][KB_REG];
#pragma unroll
  for (int j = 0; j < 2; ++j) {
    const int col = (nb0 + j) * 16 + (lane & 15);
#pragma unroll
    for (int kb = 0; kb < KB_REG; ++kb) {
      const int Kb = D_DIM + (KB_LDS + kb) * 32 + ((lane >= 16) ? 8 : 0);
      v16bf b;
#pragma unroll
      for (int e = 0; e < 16; ++e)
        b[e] = (__bf16)W[(size_t)(Kb + e + ((e >= 8) ? 8 : 0)) * H_DIM + col];
      whi[j][kb] = b;
    }
  }
  __syncthreads();

  // Index precomputation (C/D layout: M = r + 8*(lane>=16), N = lane&15;
  // hbuf scatter positions derived from the A-fragment layout).
  const int n0   = nb0 * 16 + (lane & 15);   // column of acc0 lanes
  const int mOff = (lane >= 16) ? 8 : 0;
  const int hHi  = (lane & 8) ? 16 : 0;      // lane-half of hbuf target
  const int hB0  = (n0 >> 5) * 512 + (lane & 7);   // block 0 scatter base
  const int hB1  = hB0 + 8;                        // block 1: e += 8

  for (int t = 0; t < T_DIM; ++t) {
    const int wp = t & 1;                    // write parity
    const int rp = wp ^ 1;                   // read parity (t-1's writes)
    __bf16* hW = hbuf + wp * HBUF_ELEMS;
    const __bf16* hR = hbuf + rp * HBUF_ELEMS;

    // Hoist xw_t loads (single-use -> non-temporal); each slot has exactly
    // one owner thread, so these overlap the WMMA chain safely.
    float xw0[8], xw1[8];
    const size_t oBase = ((size_t)(rowB + mOff) * T_DIM + (size_t)t) * H_DIM + n0;
#pragma unroll
    for (int r = 0; r < 8; ++r) {
      const size_t o = oBase + (size_t)r * T_DIM * H_DIM;
      xw0[r] = __builtin_nontemporal_load(&out[o]);
      xw1[r] = __builtin_nontemporal_load(&out[o + 16]);
    }

    v8f acc0 = {0.f, 0.f, 0.f, 0.f, 0.f, 0.f, 0.f, 0.f};
    v8f acc1 = {0.f, 0.f, 0.f, 0.f, 0.f, 0.f, 0.f, 0.f};
    if (t > 0) {
#pragma unroll
      for (int kb = 0; kb < KB_TOT; ++kb) {
        const v16bf a = *(const v16bf*)&hR[(kb * 32 + lane) * 16];
        v16bf b0, b1;
        if (kb < KB_LDS) {
          b0 = *(const v16bf*)&WhL[((kb * 32 + nb0) * 32 + lane) * 16];
          b1 = *(const v16bf*)&WhL[((kb * 32 + nb0 + 1) * 32 + lane) * 16];
        } else {
          b0 = whi[0][kb - KB_LDS];
          b1 = whi[1][kb - KB_LDS];
        }
        acc0 = __builtin_amdgcn_wmma_f32_16x16x32_bf16(false, a, false, b0,
                                                       (short)0, acc0, false, false);
        acc1 = __builtin_amdgcn_wmma_f32_16x16x32_bf16(false, a, false, b1,
                                                       (short)0, acc1, false, false);
      }
    }

    // h = tanh(xw + acc): fp32 out (non-temporal, never re-read) + bf16 state.
#pragma unroll
    for (int r = 0; r < 8; ++r) {
      const int    M = mOff + r;
      const size_t o = oBase + (size_t)r * T_DIM * H_DIM;
      const float v0 = fast_tanh(xw0[r] + acc0[r]);
      const float v1 = fast_tanh(xw1[r] + acc1[r]);
      __builtin_nontemporal_store(v0, &out[o]);
      __builtin_nontemporal_store(v1, &out[o + 16]);
      hW[hB0 + (M + hHi) * 16] = (__bf16)v0;
      hW[hB1 + (M + hHi) * 16] = (__bf16)v1;
    }

    // Prefetch next step's xw tile while we wait at the barrier.
    if (t + 1 < T_DIM) {
      __builtin_prefetch(
          out + ((size_t)(rowB + mOff) * T_DIM + (size_t)(t + 1)) * H_DIM + n0, 0, 1);
    }
    // Single barrier per step: publishes hW and protects hR for stragglers.
    __syncthreads();
  }
}

// ---------------------------------------------------------------------------
// Host-side launcher.  Inputs (setup_inputs order): x f32[64,1024,512],
// W f32[1024,512], b f32[512].  Output: f32[64,1024,512].  No workspace use.
// ---------------------------------------------------------------------------
extern "C" void kernel_launch(void* const* d_in, const int* in_sizes, int n_in,
                              void* d_out, int out_size, void* d_ws, size_t ws_size,
                              hipStream_t stream) {
  (void)in_sizes; (void)n_in; (void)out_size; (void)d_ws; (void)ws_size;
  const float* x    = (const float*)d_in[0];
  const float* W    = (const float*)d_in[1];
  const float* bias = (const float*)d_in[2];
  float*       out  = (float*)d_out;

  // Phase 1: xw = x @ Wx + b  ->  d_out
  dim3 g1(H_DIM / 16, 64), b1(128);
  rnn_xw_gemm<<<g1, b1, 0, stream>>>(x, W, bias, out);

  // Phase 2: barrier-free persistent scan, in place over d_out.
  dim3 g2(B_DIM / 16), b2(512);
  rnn_scan_local<<<g2, b2, SCAN_LDS_BYTES, stream>>>(W, out);
}
